// MultiHeadGraphAttention_52012053954975
// MI455X (gfx1250) — matrix operand order repaired and checked by reading, not compiled, not run on hardware
//
#include <hip/hip_runtime.h>
#include <hip/hip_bf16.h>
#include <stdint.h>

// Problem constants (match reference)
#define NN 40000
#define EE 640000
#define DD 128
#define HH 8
#define HDIM 16
#define LS 132          // padded LDS row stride (floats) to spread banks

typedef __attribute__((ext_vector_type(16))) __bf16 v16bf;
typedef __attribute__((ext_vector_type(8)))  __bf16 v8bf;
typedef __attribute__((ext_vector_type(8)))  float  v8f;

// ---------- helpers -----------------------------------------------------

// monotone float<->uint encoding so we can use integer atomicMax for floats
__device__ __forceinline__ unsigned encf(float x) {
  unsigned u = __float_as_uint(x);
  return (u & 0x80000000u) ? ~u : (u | 0x80000000u);
}
__device__ __forceinline__ float decf(unsigned e) {
  unsigned u = (e & 0x80000000u) ? (e & 0x7fffffffu) : ~e;
  return __uint_as_float(u);
}

__device__ __forceinline__ float wsum(float v) {
#pragma unroll
  for (int o = 16; o > 0; o >>= 1) v += __shfl_xor(v, o, 32);
  return v;
}

// A-fragment (16x32 bf16, MxK) element->K map per ISA table:
// lane half g: VGPR j<4 -> K = 8g+2j+h ; j>=4 -> K = 16+8g+2(j-4)+h
__device__ __forceinline__ int kmapA(int e, int g) {
  int j = e >> 1, h = e & 1;
  return (j < 4) ? (8 * g + 2 * j + h) : (16 + 8 * g + 2 * (j - 4) + h);
}
// B-fragment (32x16 bf16, KxN): K = 8g + e (e<8), 16 + 8g + (e-8) (e>=8), N = lane&15

// Convert fp32 tile in LDS (16 x 128, stride LS) into A-fragment-layout
// bf16 hi/lo LDS buffers: element (kk,lane,e) at [(kk*32+lane)*16 + e].
// 256 threads: each handles half of one (kk,lane) fragment (8 elements) and
// writes it with a single 16B vector store.
__device__ __forceinline__ void convert_frag(const float* __restrict__ sA,
                                             __bf16* __restrict__ aHi,
                                             __bf16* __restrict__ aLo,
                                             int tid) {
  const int combo = tid >> 1;       // 0..127 = (kk, lane)
  const int kk = combo >> 5;        // 0..3
  const int lane = combo & 31;
  const int g = lane >> 4, lrow = lane & 15;
  const int e0 = (tid & 1) * 8;
  v8bf hi, lo;
#pragma unroll
  for (int j = 0; j < 8; ++j) {
    int e = e0 + j;
    float xv = sA[lrow * LS + kk * 32 + kmapA(e, g)];
    __bf16 h = (__bf16)xv;
    hi[j] = h;
    lo[j] = (__bf16)(xv - (float)h);
  }
  *(v8bf*)&aHi[combo * 16 + e0] = hi;
  *(v8bf*)&aLo[combo * 16 + e0] = lo;
}

// Error-compensated bf16 GEMM on pre-swizzled fragments.
// A fragments in LDS (fragment layout), W fragments in global (fragment
// layout, pre-split hi/lo).  3 WMMAs per 32-wide K block: hh + hl + lh.
__device__ __forceinline__ v8f gemm16f(const __bf16* __restrict__ aHi,
                                       const __bf16* __restrict__ aLo,
                                       const __bf16* __restrict__ bHi,
                                       const __bf16* __restrict__ bLo,
                                       int tile, int lane, v8f acc) {
  const v16bf* av_h = (const v16bf*)aHi;
  const v16bf* av_l = (const v16bf*)aLo;
  const v16bf* bv_h = (const v16bf*)bHi;
  const v16bf* bv_l = (const v16bf*)bLo;
#pragma unroll
  for (int kk = 0; kk < 4; ++kk) {
    v16bf ah = av_h[kk * 32 + lane];
    v16bf al = av_l[kk * 32 + lane];
    v16bf bh = bv_h[(tile * 4 + kk) * 32 + lane];
    v16bf bl = bv_l[(tile * 4 + kk) * 32 + lane];
    acc = __builtin_amdgcn_wmma_f32_16x16x32_bf16(false, ah, false, bh,
                                                  (short)0, acc, false, false);
    acc = __builtin_amdgcn_wmma_f32_16x16x32_bf16(false, ah, false, bl,
                                                  (short)0, acc, false, false);
    acc = __builtin_amdgcn_wmma_f32_16x16x32_bf16(false, al, false, bh,
                                                  (short)0, acc, false, false);
  }
  return acc;
}

// ---------- kernels -----------------------------------------------------

// Split one 128x128 fp32 weight matrix into bf16 hi/lo copies, stored in
// B-fragment layout: t = (((tile*4 + kk)*32 + lane)*16 + e),
// K = kk*32 + (e<8 ? 8g+e : 16+8g+e-8), g = lane>>4, col = tile*16 + (lane&15).
__global__ void wprep_kernel(const float* __restrict__ W,
                             __bf16* __restrict__ hi, __bf16* __restrict__ lo) {
  int t = blockIdx.x * 256 + threadIdx.x;
  if (t >= DD * DD) return;
  int e = t & 15;
  int lane = (t >> 4) & 31;
  int kk = (t >> 9) & 3;
  int tile = t >> 11;
  int g = lane >> 4, lrow = lane & 15;
  int K = kk * 32 + ((e < 8) ? (8 * g + e) : (16 + 8 * g + (e - 8)));
  int col = tile * 16 + lrow;
  float x = W[K * DD + col];
  __bf16 h = (__bf16)x;
  hi[t] = h;
  lo[t] = (__bf16)(x - (float)h);
}

__global__ void init_kernel(float* __restrict__ x_out,
                            float* __restrict__ denom,
                            unsigned* __restrict__ nmax) {
  int t = blockIdx.x * 256 + threadIdx.x;
  if (t < NN * DD) x_out[t] = 0.f;
  if (t < NN * HH) {
    denom[t] = 0.f;
    nmax[t] = encf(-1e30f);
  }
}

// q = x@Wq + bq ; k = x@Wk ; v = x@Wv   (16 rows/block, 8 waves = 8 col tiles)
__global__ void __launch_bounds__(256, 1)
qkv_kernel(const float* __restrict__ x,
           const __bf16* __restrict__ qh, const __bf16* __restrict__ ql,
           const float* __restrict__ wq_b,
           const __bf16* __restrict__ kh, const __bf16* __restrict__ kl,
           const __bf16* __restrict__ vh, const __bf16* __restrict__ vl,
           float* __restrict__ q, float* __restrict__ k, float* __restrict__ v) {
  __shared__ float sA[16 * LS];
  __shared__ alignas(32) __bf16 aHi[2048];
  __shared__ alignas(32) __bf16 aLo[2048];
  const int tid = threadIdx.x, wave = tid >> 5, lane = tid & 31;
  const int g = lane >> 4, lrow = lane & 15;
  const int row0 = blockIdx.x * 16;
  for (int i = tid; i < 16 * DD; i += 256) {
    int r = i >> 7, c = i & 127;
    sA[r * LS + c] = x[(size_t)(row0 + r) * DD + c];
  }
  __syncthreads();
  convert_frag(sA, aHi, aLo, tid);
  __syncthreads();
  const int c0 = wave * 16;
  v8f z = {0.f, 0.f, 0.f, 0.f, 0.f, 0.f, 0.f, 0.f};
  v8f aq = gemm16f(aHi, aLo, qh, ql, wave, lane, z);
  v8f ak = gemm16f(aHi, aLo, kh, kl, wave, lane, z);
  v8f av = gemm16f(aHi, aLo, vh, vl, wave, lane, z);
  float bq = wq_b[c0 + lrow];
#pragma unroll
  for (int r = 0; r < 8; ++r) {
    size_t row = (size_t)(row0 + r + 8 * g);
    q[row * DD + c0 + lrow] = aq[r] + bq;
    k[row * DD + c0 + lrow] = ak[r];
    v[row * DD + c0 + lrow] = av[r];
  }
}

// Fused edge kernel: eb/ew GEMMs, score, att logits (+atomic max),
// weo GEMM + residual + LayerNorm -> e_out.  16 edges per block.
__global__ void __launch_bounds__(256, 1)
edge_kernel(const float* __restrict__ edge_attr, const int* __restrict__ ei,
            const float* __restrict__ q, const float* __restrict__ k,
            const __bf16* __restrict__ ebh, const __bf16* __restrict__ ebl,
            const float* __restrict__ w_eb_b,
            const __bf16* __restrict__ ewh, const __bf16* __restrict__ ewl,
            const float* __restrict__ w_ew_b,
            const __bf16* __restrict__ weoh, const __bf16* __restrict__ weol,
            const float* __restrict__ Aw,
            const float* __restrict__ ln_g, const float* __restrict__ ln_b,
            float* __restrict__ att, unsigned* __restrict__ nmax,
            float* __restrict__ e_out) {
  __shared__ float sIn[16 * LS];  // edge_attr tile
  __shared__ float sT0[16 * LS];  // eb -> score
  __shared__ float sT1[16 * LS];  // ew -> (weo out + residual)
  __shared__ alignas(32) __bf16 aHi[2048];
  __shared__ alignas(32) __bf16 aLo[2048];
  __shared__ int sSrc[16], sDst[16];
  const int tid = threadIdx.x, wave = tid >> 5, lane = tid & 31;
  const int g = lane >> 4, lrow = lane & 15;
  const int e0 = blockIdx.x * 16;
  if (tid < 16) {
    sSrc[tid] = ei[e0 + tid];
    sDst[tid] = ei[EE + e0 + tid];
  }
  for (int i = tid; i < 16 * DD; i += 256) {
    int r = i >> 7, c = i & 127;
    sIn[r * LS + c] = edge_attr[(size_t)(e0 + r) * DD + c];
  }
  __syncthreads();
  convert_frag(sIn, aHi, aLo, tid);
  __syncthreads();
  const int c0 = wave * 16;
  v8f z = {0.f, 0.f, 0.f, 0.f, 0.f, 0.f, 0.f, 0.f};
  {
    v8f eb = gemm16f(aHi, aLo, ebh, ebl, wave, lane, z);
    v8f ew = gemm16f(aHi, aLo, ewh, ewl, wave, lane, z);
    float bb = w_eb_b[c0 + lrow];
    float bw = w_ew_b[c0 + lrow];
#pragma unroll
    for (int r = 0; r < 8; ++r) {
      int row = r + 8 * g;
      sT0[row * LS + c0 + lrow] = eb[r] + bb;
      sT1[row * LS + c0 + lrow] = ew[r] + bw;
    }
  }
  __syncthreads();
  // score = relu(signed_sqrt((k[src]+q[dst])*ew) + eb)   (elementwise, in place)
  for (int i = tid; i < 16 * DD; i += 256) {
    int r = i >> 7, c = i & 127;
    float gsum = k[(size_t)sSrc[r] * DD + c] + q[(size_t)sDst[r] * DD + c];
    float s = gsum * sT1[r * LS + c];
    s = copysignf(sqrtf(fabsf(s)), s);
    s = fmaxf(s + sT0[r * LS + c], 0.f);
    sT0[r * LS + c] = s;
  }
  __syncthreads();
  // re-swizzle score into A fragments (aHi/aLo free again after eb/ew GEMMs)
  convert_frag(sT0, aHi, aLo, tid);
  // attention logits per (edge, head), clip, scatter-max into nmax
  if (tid < 128) {
    int e = tid >> 3, hh = tid & 7;
    float a = 0.f;
#pragma unroll
    for (int d = 0; d < HDIM; ++d) a += sT0[e * LS + hh * HDIM + d] * Aw[d * HH + hh];
    a = fminf(fmaxf(a, -5.0f), 5.0f);
    att[(size_t)(e0 + e) * HH + hh] = a;
    atomicMax(&nmax[(size_t)sDst[e] * HH + hh], encf(a));
  }
  __syncthreads();
  // e_out = LN(score @ weo + edge_attr)
  {
    v8f o = gemm16f(aHi, aLo, weoh, weol, wave, lane, z);
#pragma unroll
    for (int r = 0; r < 8; ++r) {
      int row = r + 8 * g;
      sT1[row * LS + c0 + lrow] = o[r] + sIn[row * LS + c0 + lrow];
    }
  }
  __syncthreads();
#pragma unroll
  for (int rr = 0; rr < 2; ++rr) {
    int r = wave * 2 + rr;
    float s = 0.f, s2 = 0.f;
#pragma unroll
    for (int j = 0; j < 4; ++j) {
      float xv = sT1[r * LS + lane * 4 + j];
      s += xv;
      s2 += xv * xv;
    }
    s = wsum(s);
    s2 = wsum(s2);
    float mean = s * (1.0f / DD);
    float var = s2 * (1.0f / DD) - mean * mean;
    float rstd = rsqrtf(var + 1e-5f);
#pragma unroll
    for (int j = 0; j < 4; ++j) {
      int c = lane * 4 + j;
      e_out[(size_t)(e0 + r) * DD + c] =
          (sT1[r * LS + c] - mean) * rstd * ln_g[c] + ln_b[c];
    }
  }
}

__global__ void exp_kernel(const int* __restrict__ ei,
                           const unsigned* __restrict__ nmax,
                           float* __restrict__ att, float* __restrict__ denom) {
  int t = blockIdx.x * 256 + threadIdx.x;
  if (t >= EE * HH) return;
  int e = t >> 3, hh = t & 7;
  int dst = ei[EE + e];
  float m = decf(nmax[(size_t)dst * HH + hh]);
  float p = __expf(att[t] - m);
  att[t] = p;
  atomicAdd(&denom[(size_t)dst * HH + hh], p);
}

__global__ void scatter_kernel(const int* __restrict__ ei,
                               const float* __restrict__ v,
                               const float* __restrict__ att,
                               const float* __restrict__ denom,
                               float* __restrict__ x_out) {
  long long t = (long long)blockIdx.x * 256 + threadIdx.x;
  if (t >= (long long)EE * DD) return;
  int e = (int)(t >> 7), c = (int)(t & 127);
  int hh = c >> 4;
  int src = ei[e], dst = ei[EE + e];
  float w = att[(size_t)e * HH + hh] / (denom[(size_t)dst * HH + hh] + 1e-16f);
  atomicAdd(&x_out[(size_t)dst * DD + c], v[(size_t)src * DD + c] * w);
}

// h = LN(x_out @ wo + x)
__global__ void __launch_bounds__(256, 1)
nodeout_kernel(const float* __restrict__ x_acc, const float* __restrict__ x,
               const __bf16* __restrict__ woh, const __bf16* __restrict__ wol,
               const float* __restrict__ ln_g, const float* __restrict__ ln_b,
               float* __restrict__ outh) {
  __shared__ float sA[16 * LS];
  __shared__ float sB[16 * LS];
  __shared__ alignas(32) __bf16 aHi[2048];
  __shared__ alignas(32) __bf16 aLo[2048];
  const int tid = threadIdx.x, wave = tid >> 5, lane = tid & 31;
  const int g = lane >> 4, lrow = lane & 15;
  const int row0 = blockIdx.x * 16;
  for (int i = tid; i < 16 * DD; i += 256) {
    int r = i >> 7, c = i & 127;
    sA[r * LS + c] = x_acc[(size_t)(row0 + r) * DD + c];
  }
  __syncthreads();
  convert_frag(sA, aHi, aLo, tid);
  __syncthreads();
  const int c0 = wave * 16;
  v8f z = {0.f, 0.f, 0.f, 0.f, 0.f, 0.f, 0.f, 0.f};
  v8f o = gemm16f(aHi, aLo, woh, wol, wave, lane, z);
#pragma unroll
  for (int r = 0; r < 8; ++r) {
    int row = r + 8 * g;
    sB[row * LS + c0 + lrow] = o[r] + x[(size_t)(row0 + row) * DD + c0 + lrow];
  }
  __syncthreads();
#pragma unroll
  for (int rr = 0; rr < 2; ++rr) {
    int r = wave * 2 + rr;
    float s = 0.f, s2 = 0.f;
#pragma unroll
    for (int j = 0; j < 4; ++j) {
      float xv = sB[r * LS + lane * 4 + j];
      s += xv;
      s2 += xv * xv;
    }
    s = wsum(s);
    s2 = wsum(s2);
    float mean = s * (1.0f / DD);
    float var = s2 * (1.0f / DD) - mean * mean;
    float rstd = rsqrtf(var + 1e-5f);
#pragma unroll
    for (int j = 0; j < 4; ++j) {
      int c = lane * 4 + j;
      outh[(size_t)(row0 + r) * DD + c] =
          (sB[r * LS + c] - mean) * rstd * ln_g[c] + ln_b[c];
    }
  }
}

// ---------- launch ------------------------------------------------------

extern "C" void kernel_launch(void* const* d_in, const int* in_sizes, int n_in,
                              void* d_out, int out_size, void* d_ws,
                              size_t ws_size, hipStream_t stream) {
  const float* x         = (const float*)d_in[0];
  const int*   ei        = (const int*)d_in[1];   // (2,E), row0=src, row1=dst
  const float* edge_attr = (const float*)d_in[2];
  const float* wq_w  = (const float*)d_in[3];
  const float* wq_b  = (const float*)d_in[4];
  const float* wk_w  = (const float*)d_in[5];
  const float* wv_w  = (const float*)d_in[6];
  const float* web_w = (const float*)d_in[7];
  const float* web_b = (const float*)d_in[8];
  const float* wew_w = (const float*)d_in[9];
  const float* wew_b = (const float*)d_in[10];
  const float* wo_w  = (const float*)d_in[11];
  const float* weo_w = (const float*)d_in[12];
  const float* Aw    = (const float*)d_in[13];
  const float* ln_n_g = (const float*)d_in[14];
  const float* ln_n_b = (const float*)d_in[15];
  const float* ln_e_g = (const float*)d_in[16];
  const float* ln_e_b = (const float*)d_in[17];

  float* outh = (float*)d_out;
  float* oute = outh + (size_t)NN * DD;

  // workspace layout (~105 MB fp32 + 0.5 MB bf16 weight fragments)
  float* ws    = (float*)d_ws;
  float* q     = ws;
  float* k     = q + (size_t)NN * DD;
  float* v     = k + (size_t)NN * DD;
  float* x_out = v + (size_t)NN * DD;
  float* att   = x_out + (size_t)NN * DD;            // E*H
  float* denom = att + (size_t)EE * HH;              // N*H
  unsigned* nmax = (unsigned*)(denom + (size_t)NN * HH);
  __bf16* wb = (__bf16*)(nmax + (size_t)NN * HH);
  __bf16* qh  = wb;                __bf16* ql  = qh  + DD * DD;
  __bf16* kh  = ql  + DD * DD;     __bf16* kl  = kh  + DD * DD;
  __bf16* vh  = kl  + DD * DD;     __bf16* vl  = vh  + DD * DD;
  __bf16* ebh = vl  + DD * DD;     __bf16* ebl = ebh + DD * DD;
  __bf16* ewh = ebl + DD * DD;     __bf16* ewl = ewh + DD * DD;
  __bf16* woh = ewl + DD * DD;     __bf16* wol = woh + DD * DD;
  __bf16* weoh = wol + DD * DD;    __bf16* weol = weoh + DD * DD;

  const int WB = (DD * DD + 255) / 256;
  wprep_kernel<<<WB, 256, 0, stream>>>(wq_w, qh, ql);
  wprep_kernel<<<WB, 256, 0, stream>>>(wk_w, kh, kl);
  wprep_kernel<<<WB, 256, 0, stream>>>(wv_w, vh, vl);
  wprep_kernel<<<WB, 256, 0, stream>>>(web_w, ebh, ebl);
  wprep_kernel<<<WB, 256, 0, stream>>>(wew_w, ewh, ewl);
  wprep_kernel<<<WB, 256, 0, stream>>>(wo_w, woh, wol);
  wprep_kernel<<<WB, 256, 0, stream>>>(weo_w, weoh, weol);

  init_kernel<<<(NN * DD + 255) / 256, 256, 0, stream>>>(x_out, denom, nmax);

  qkv_kernel<<<NN / 16, 256, 0, stream>>>(x, qh, ql, wq_b, kh, kl, vh, vl,
                                          q, k, v);

  edge_kernel<<<EE / 16, 256, 0, stream>>>(edge_attr, ei, q, k,
                                           ebh, ebl, web_b, ewh, ewl, wew_b,
                                           weoh, weol, Aw, ln_e_g, ln_e_b,
                                           att, nmax, oute);

  exp_kernel<<<(EE * HH + 255) / 256, 256, 0, stream>>>(ei, nmax, att, denom);

  scatter_kernel<<<(int)(((long long)EE * DD + 255) / 256), 256, 0, stream>>>(
      ei, v, att, denom, x_out);

  nodeout_kernel<<<NN / 16, 256, 0, stream>>>(x_out, x, woh, wol,
                                              ln_n_g, ln_n_b, outh);
}
